// GroupedQueryAttention_4346506904104
// MI455X (gfx1250) — compile-verified, matrix operand order checked by reading, not run on hardware
//
#include <hip/hip_runtime.h>
#include <hip/hip_bf16.h>
#include <stddef.h>
#include <stdint.h>

// Problem constants (match reference)
#define BB   2
#define SS   2048
#define DD   2048
#define HH   32
#define KVHH 8
#define DHH  64
#define NREP 4          // HH / KVHH
#define NEGF (-1e30f)

typedef __attribute__((ext_vector_type(16))) __bf16 v16bf;
typedef __attribute__((ext_vector_type(8)))  __bf16 v8bf;
typedef __attribute__((ext_vector_type(8)))  float  v8f;

// ---------------------------------------------------------------------------
// WMMA fragment helpers (v_wmma_f32_16x16x32_bf16):
// A/B: lane holds row/col (lane&15); element e -> K = (e&7)+8*(lane>>4)+16*(e>>3)
// C/D: lane holds col (lane&15); vgpr r -> row r + 8*(lane>>4).
// Fragment = two contiguous 16B loads at koff and koff+16.
// ---------------------------------------------------------------------------
__device__ __forceinline__ v16bf load_frag(const __bf16* base, int koff) {
  v8bf lo = *(const v8bf*)(base + koff);
  v8bf hi = *(const v8bf*)(base + koff + 16);
  return __builtin_shufflevector(lo, hi, 0,1,2,3,4,5,6,7,8,9,10,11,12,13,14,15);
}

__device__ __forceinline__ v8f wmma_bf16(v16bf a, v16bf b, v8f c) {
  return __builtin_amdgcn_wmma_f32_16x16x32_bf16(false, a, false, b, (short)0, c,
                                                 false, false);
}

// Async global -> LDS DMA (ASYNCcnt tracked), 16B per lane, GVS addressing.
__device__ __forceinline__ void async_b128(unsigned lds_addr,
                                           const __bf16* gbase, int voff_bytes) {
  asm volatile("global_load_async_to_lds_b128 %0, %1, %2"
               :: "v"(lds_addr), "v"(voff_bytes),
                  "s"((unsigned long long)(uintptr_t)gbase)
               : "memory");
}

// ---------------------------------------------------------------------------
// Kernel 0: fp32 -> bf16 bulk conversion (one pass; makes all GEMM operands
// async-DMA-able as bf16).
// ---------------------------------------------------------------------------
__global__ __launch_bounds__(256)
void cvt_bf16_kernel(const float* __restrict__ in, __bf16* __restrict__ out,
                     int n8) {
  int i = blockIdx.x * 256 + threadIdx.x;
  if (i >= n8) return;
  const float4* p = (const float4*)in + (size_t)i * 2;
  float4 a = p[0], b = p[1];
  v8bf o;
  o[0] = (__bf16)a.x; o[1] = (__bf16)a.y; o[2] = (__bf16)a.z; o[3] = (__bf16)a.w;
  o[4] = (__bf16)b.x; o[5] = (__bf16)b.y; o[6] = (__bf16)b.z; o[7] = (__bf16)b.w;
  *(v8bf*)(out + (size_t)i * 8) = o;
}

// ---------------------------------------------------------------------------
// Shared 128x128 GEMM mainloop (K = 2048, chunks of 32), double-buffered
// async global->LDS staging. Block = 256 threads (8 waves, 4M x 2N); each
// wave computes 32x64 (2 A-frags x 4 B-frags -> 8 WMMAs per chunk).
// LDS layout (bytes): [A buf0 8K][A buf1 8K][B buf0 8K][B buf1 8K] = 32KB.
// ---------------------------------------------------------------------------
__device__ __forceinline__ void gemm_128x128(const __bf16* __restrict__ Ab,  // + m0*DD
                                             const __bf16* __restrict__ Bb,  // + n0*DD
                                             __bf16* sm, v8f acc[2][4]) {
  const int tid  = threadIdx.x;
  const int lane = tid & 31;
  const int wave = tid >> 5;
  const int half = (lane >> 4) & 1;
  const int lx   = lane & 15;
  const int koff = half * 8;
  const int wm   = wave & 3;
  const int wn   = wave >> 2;

  const unsigned lds0 = (unsigned)(uintptr_t)sm;   // LDS byte offset of sm
  // Per-thread DMA slots: 512 x 16B chunks per 8KB tile, 2 per thread.
  const int r0 = tid >> 2;        // row 0..63 (+64 for slot 1)
  const int c0 = tid & 3;         // 16B chunk within 64B row
  int gOff[2], lOff[2];
#pragma unroll
  for (int i = 0; i < 2; ++i) {
    gOff[i] = (r0 + 64 * i) * (DD * 2) + c0 * 16;  // bytes in global (row stride 4KB)
    lOff[i] = (r0 + 64 * i) * 64 + c0 * 16;        // bytes in 8KB LDS tile
  }

  // Prologue: stage tile 0 into buffer 0.
#pragma unroll
  for (int i = 0; i < 2; ++i) async_b128(lds0 + lOff[i], Ab, gOff[i]);
#pragma unroll
  for (int i = 0; i < 2; ++i) async_b128(lds0 + 16384u + lOff[i], Bb, gOff[i]);

  for (int kk = 0; kk < DD / 32; ++kk) {
    const int cur = kk & 1;
    if (kk < DD / 32 - 1) {
      // Kick off next tile's DMA into the other buffer, then wait for the
      // current tile (4 newer ops may remain in flight).
      const int      kb   = (kk + 1) * 64;             // 32 bf16 = 64 bytes
      const unsigned bufb = (cur ^ 1) ? 8192u : 0u;
#pragma unroll
      for (int i = 0; i < 2; ++i)
        async_b128(lds0 + bufb + lOff[i], Ab, gOff[i] + kb);
#pragma unroll
      for (int i = 0; i < 2; ++i)
        async_b128(lds0 + 16384u + bufb + lOff[i], Bb, gOff[i] + kb);
      asm volatile("s_wait_asynccnt 4" ::: "memory");
    } else {
      asm volatile("s_wait_asynccnt 0" ::: "memory");
    }
    __syncthreads();

    const __bf16* As = sm + cur * 4096;
    const __bf16* Bs = sm + 8192 + cur * 4096;
    v16bf aF[2], bF[4];
#pragma unroll
    for (int mi = 0; mi < 2; ++mi)
      aF[mi] = load_frag(As + (wm * 32 + mi * 16 + lx) * 32, koff);
#pragma unroll
    for (int t = 0; t < 4; ++t)
      bF[t] = load_frag(Bs + (wn * 64 + t * 16 + lx) * 32, koff);
#pragma unroll
    for (int mi = 0; mi < 2; ++mi)
#pragma unroll
      for (int t = 0; t < 4; ++t)
        acc[mi][t] = wmma_bf16(aF[mi], bF[t], acc[mi][t]);
    __syncthreads();   // protect buffer (cur) before it is re-filled
  }
}

// ---------------------------------------------------------------------------
// Kernel 1: fused QKV projection + RoPE.
// C(4096 x 3072) = xb @ [Wq;Wk;Wv]^T, RoPE epilogue via shfl_xor(.,1) pairing.
// Writes Q head-major [b][h][s][dh], K head-major, V transposed [b][kvh][dh][s].
// ---------------------------------------------------------------------------
__global__ __launch_bounds__(256)
void qkv_rope_kernel(const __bf16* __restrict__ xb,
                     const __bf16* __restrict__ Wqkvb,
                     const float* __restrict__ cosb,
                     const float* __restrict__ sinb,
                     __bf16* __restrict__ Qws,
                     __bf16* __restrict__ Kws,
                     __bf16* __restrict__ Vt) {
  __shared__ alignas(16) __bf16 sm[16384];
  const int m0 = blockIdx.y * 128;
  const int n0 = blockIdx.x * 128;

  v8f acc[2][4] = {};
  gemm_128x128(xb + (size_t)m0 * DD, Wqkvb + (size_t)n0 * DD, sm, acc);

  const int tid  = threadIdx.x;
  const int lane = tid & 31;
  const int wave = tid >> 5;
  const int half = (lane >> 4) & 1;
  const int lx   = lane & 15;
  const int wm   = wave & 3;
  const int wn   = wave >> 2;

#pragma unroll
  for (int mi = 0; mi < 2; ++mi) {
#pragma unroll
    for (int t = 0; t < 4; ++t) {
#pragma unroll
      for (int r = 0; r < 8; ++r) {
        float v   = acc[mi][t][r];
        float oth = __shfl_xor(v, 1, 32);      // RoPE pair partner (adj column)
        int   col = n0 + wn * 64 + t * 16 + lx;
        int   row = m0 + wm * 32 + mi * 16 + r + half * 8;
        int   b   = row >> 11;
        int   s_  = row & (SS - 1);
        if (col < 2048) {                      // Q, rope
          int h = col >> 6, dh = col & 63;
          float c  = cosb[s_ * 32 + (dh >> 1)];
          float sn = sinb[s_ * 32 + (dh >> 1)];
          float o  = ((dh & 1) == 0) ? (v * c - oth * sn) : (oth * sn + v * c);
          Qws[(((size_t)b * HH + h) * SS + s_) * DHH + dh] = (__bf16)o;
        } else if (col < 2560) {               // K, rope
          int j = col - 2048;
          int kh = j >> 6, dh = j & 63;
          float c  = cosb[s_ * 32 + (dh >> 1)];
          float sn = sinb[s_ * 32 + (dh >> 1)];
          float o  = ((dh & 1) == 0) ? (v * c - oth * sn) : (oth * sn + v * c);
          Kws[(((size_t)b * KVHH + kh) * SS + s_) * DHH + dh] = (__bf16)o;
        } else {                               // V, transposed [dh][s]
          int j = col - 2560;
          int kh = j >> 6, dh = j & 63;
          Vt[(((size_t)b * KVHH + kh) * DHH + dh) * SS + s_] = (__bf16)v;
        }
      }
    }
  }
}

// ---------------------------------------------------------------------------
// Kernel 2: causal flash attention. Block = 128 threads (4 waves); wave owns
// 16 query rows of one head. Per 64-key tile: QK^T (8 WMMAs), online softmax
// (16-lane half-wave reductions matching C layout), P staged through LDS
// (C->A layout transpose), P*V (8 WMMAs; V pre-transposed for contiguous
// B-fragments).
// ---------------------------------------------------------------------------
__global__ __launch_bounds__(128)
void attn_kernel(const __bf16* __restrict__ Qws,
                 const __bf16* __restrict__ Kws,
                 const __bf16* __restrict__ Vt,
                 __bf16* __restrict__ Ows) {
  __shared__ alignas(16) __bf16 Pw[4][16][64];

  const int tid  = threadIdx.x;
  const int lane = tid & 31;
  const int wave = tid >> 5;
  const int half = (lane >> 4) & 1;
  const int lx   = lane & 15;
  const int koff = half * 8;

  const int h   = blockIdx.y % HH;
  const int b   = blockIdx.y / HH;
  const int kvh = h / NREP;
  const int q0  = blockIdx.x * 64 + wave * 16;

  const __bf16* Qbase = Qws + ((size_t)b * HH + h) * SS * DHH;
  const __bf16* Kbase = Kws + ((size_t)b * KVHH + kvh) * SS * DHH;
  const __bf16* Vbase = Vt  + ((size_t)b * KVHH + kvh) * DHH * SS;

  v16bf aQ[2];
#pragma unroll
  for (int c = 0; c < 2; ++c)
    aQ[c] = load_frag(Qbase + (size_t)(q0 + lx) * DHH + c * 32, koff);

  float rowmax[8], rowsum[8];
#pragma unroll
  for (int r = 0; r < 8; ++r) { rowmax[r] = NEGF; rowsum[r] = 0.f; }
  v8f oacc[4] = {};
  const float scale = 0.125f;          // DH^-0.5

  const int ktmax = (q0 + 15) >> 6;
  for (int kt = 0; kt <= ktmax; ++kt) {
    const int k0 = kt * 64;

    v8f sacc[4] = {};
#pragma unroll
    for (int t = 0; t < 4; ++t) {
#pragma unroll
      for (int c = 0; c < 2; ++c) {
        v16bf bK = load_frag(Kbase + (size_t)(k0 + t * 16 + lx) * DHH + c * 32, koff);
        sacc[t] = wmma_bf16(aQ[c], bK, sacc[t]);
      }
    }

    float nmax[8];
#pragma unroll
    for (int r = 0; r < 8; ++r) {
      const int q = q0 + r + half * 8;
      float loc = NEGF;
#pragma unroll
      for (int t = 0; t < 4; ++t) {
        int   kk = k0 + t * 16 + lx;
        float sv = sacc[t][r] * scale;
        sv = (kk <= q) ? sv : NEGF;
        sacc[t][r] = sv;
        loc = fmaxf(loc, sv);
      }
      loc = fmaxf(loc, __shfl_xor(loc, 1, 32));
      loc = fmaxf(loc, __shfl_xor(loc, 2, 32));
      loc = fmaxf(loc, __shfl_xor(loc, 4, 32));
      loc = fmaxf(loc, __shfl_xor(loc, 8, 32));
      nmax[r] = fmaxf(rowmax[r], loc);
    }

#pragma unroll
    for (int r = 0; r < 8; ++r) {
      float alpha = __expf(rowmax[r] - nmax[r]);
      rowmax[r] = nmax[r];
      float rsum = 0.f;
#pragma unroll
      for (int t = 0; t < 4; ++t) {
        float p = __expf(sacc[t][r] - nmax[r]);
        rsum += p;
        Pw[wave][r + half * 8][t * 16 + lx] = (__bf16)p;
      }
      rsum += __shfl_xor(rsum, 1, 32);
      rsum += __shfl_xor(rsum, 2, 32);
      rsum += __shfl_xor(rsum, 4, 32);
      rsum += __shfl_xor(rsum, 8, 32);
      rowsum[r] = rowsum[r] * alpha + rsum;
#pragma unroll
      for (int t = 0; t < 4; ++t) oacc[t][r] *= alpha;
    }

    // same-wave LDS store->load ordering before A-fragment reads of P
    asm volatile("s_wait_dscnt 0" ::: "memory");

#pragma unroll
    for (int c = 0; c < 2; ++c) {
      v16bf aP = load_frag(&Pw[wave][lx][c * 32], koff);
#pragma unroll
      for (int t = 0; t < 4; ++t) {
        v16bf bV = load_frag(Vbase + (size_t)(t * 16 + lx) * SS + k0 + c * 32, koff);
        oacc[t] = wmma_bf16(aP, bV, oacc[t]);
      }
    }
  }

#pragma unroll
  for (int t = 0; t < 4; ++t) {
#pragma unroll
    for (int r = 0; r < 8; ++r) {
      int q  = q0 + r + half * 8;
      int dh = t * 16 + lx;
      Ows[(((size_t)b * SS + q) * HH + h) * DHH + dh] =
          (__bf16)(oacc[t][r] / rowsum[r]);
    }
  }
}

// ---------------------------------------------------------------------------
// Kernel 3: output projection. out(4096x2048 fp32) = O(bf16) @ Wo^T(bf16).
// ---------------------------------------------------------------------------
__global__ __launch_bounds__(256)
void oproj_kernel(const __bf16* __restrict__ Ows,
                  const __bf16* __restrict__ Wob,
                  float* __restrict__ out) {
  __shared__ alignas(16) __bf16 sm[16384];
  const int m0 = blockIdx.y * 128;
  const int n0 = blockIdx.x * 128;

  v8f acc[2][4] = {};
  gemm_128x128(Ows + (size_t)m0 * DD, Wob + (size_t)n0 * DD, sm, acc);

  const int tid  = threadIdx.x;
  const int lane = tid & 31;
  const int wave = tid >> 5;
  const int half = (lane >> 4) & 1;
  const int lx   = lane & 15;
  const int wm   = wave & 3;
  const int wn   = wave >> 2;

#pragma unroll
  for (int mi = 0; mi < 2; ++mi) {
#pragma unroll
    for (int t = 0; t < 4; ++t) {
#pragma unroll
      for (int r = 0; r < 8; ++r) {
        int col = n0 + wn * 64 + t * 16 + lx;
        int row = m0 + wm * 32 + mi * 16 + r + half * 8;
        out[(size_t)row * DD + col] = acc[mi][t][r];
      }
    }
  }
}

// ---------------------------------------------------------------------------
extern "C" void kernel_launch(void* const* d_in, const int* in_sizes, int n_in,
                              void* d_out, int out_size, void* d_ws, size_t ws_size,
                              hipStream_t stream) {
  (void)in_sizes; (void)n_in; (void)out_size; (void)ws_size;
  const float* x    = (const float*)d_in[0];
  const float* Wq   = (const float*)d_in[1];
  const float* Wk   = (const float*)d_in[2];
  const float* Wv   = (const float*)d_in[3];
  const float* Wo   = (const float*)d_in[4];
  const float* cosb = (const float*)d_in[5];
  const float* sinb = (const float*)d_in[6];
  float* out = (float*)d_out;

  // Workspace layout (bf16 elements)
  __bf16* xb    = (__bf16*)d_ws;                       //  8388608  (x)
  __bf16* Wqkvb = xb    + (size_t)8388608;             //  6291456  ([Wq;Wk;Wv])
  __bf16* Wob   = Wqkvb + (size_t)6291456;             //  4194304  (Wo)
  __bf16* Qws   = Wob   + (size_t)4194304;             //  8388608
  __bf16* Kws   = Qws   + (size_t)8388608;             //  2097152
  __bf16* Vt    = Kws   + (size_t)2097152;             //  2097152
  __bf16* Ows   = Vt    + (size_t)2097152;             //  8388608

  // 0) one-pass fp32 -> bf16 conversions (x, Wq|Wk|Wv concatenated, Wo)
  cvt_bf16_kernel<<<4096, 256, 0, stream>>>(x,  xb,    1048576);
  cvt_bf16_kernel<<<2048, 256, 0, stream>>>(Wq, Wqkvb,            524288);
  cvt_bf16_kernel<<< 512, 256, 0, stream>>>(Wk, Wqkvb + 4194304,  131072);
  cvt_bf16_kernel<<< 512, 256, 0, stream>>>(Wv, Wqkvb + 5242880,  131072);
  cvt_bf16_kernel<<<2048, 256, 0, stream>>>(Wo, Wob,              524288);

  // 1) fused QKV projection + RoPE: (4096 x 3072), 128x128 tiles
  qkv_rope_kernel<<<dim3(3072 / 128, (BB * SS) / 128), 256, 0, stream>>>(
      xb, Wqkvb, cosb, sinb, Qws, Kws, Vt);

  // 2) causal flash attention: (S/64 q-tiles) x (B*H heads)
  attn_kernel<<<dim3(SS / 64, BB * HH), 128, 0, stream>>>(Qws, Kws, Vt, Ows);

  // 3) output projection: (4096 x 2048), 128x128 tiles
  oproj_kernel<<<dim3(DD / 128, (BB * SS) / 128), 256, 0, stream>>>(Ows, Wob, out);
}
